// DetectionBranch_4355096838862
// MI455X (gfx1250) — matrix operand order repaired and checked by reading, not compiled, not run on hardware
//
#include <hip/hip_runtime.h>
#include <cmath>

typedef __attribute__((ext_vector_type(16))) _Float16 v16h;
typedef __attribute__((ext_vector_type(8)))  _Float16 v8h;
typedef __attribute__((ext_vector_type(8)))  float    v8f;

#define EPS_F      1e-6f
#define STRIDE_F   4.0f
#define LAMBDA_BOX 0.1f

// ---------------------------------------------------------------------------
// Kernel 0: zero the scalar loss accumulator (d_out[H*W]).
// ---------------------------------------------------------------------------
__global__ void det_init_loss(float* loss) { *loss = 0.0f; }

// ---------------------------------------------------------------------------
// Kernel 1: box head. Computes centers (written to d_out), s_hat/o_hat, and
// the L1 box+offset loss, reduced wave32-wide and atomically accumulated.
// ---------------------------------------------------------------------------
__global__ void det_box_head(const float* __restrict__ boxes,
                             const float* __restrict__ s,
                             const float* __restrict__ o,
                             float* __restrict__ centers,
                             float* __restrict__ loss,
                             int N) {
  int n = blockIdx.x * blockDim.x + threadIdx.x;
  float l = 0.0f;
  if (n < N) {
    float x1 = boxes[4 * n + 0], y1 = boxes[4 * n + 1];
    float x2 = boxes[4 * n + 2], y2 = boxes[4 * n + 3];
    float cx = 0.5f * (x1 + x2);
    float cy = 0.5f * (y1 + y2);
    centers[2 * n + 0] = cx;
    centers[2 * n + 1] = cy;
    float shx = x2 - x1, shy = y2 - y1;
    float ohx = cx / STRIDE_F - floorf(cx / STRIDE_F);
    float ohy = cy / STRIDE_F - floorf(cy / STRIDE_F);
    l = fabsf(o[2 * n + 0] - ohx) + fabsf(o[2 * n + 1] - ohy) +
        LAMBDA_BOX * (fabsf(s[2 * n + 0] - shx) + fabsf(s[2 * n + 1] - shy));
  }
  // wave32 butterfly reduction
  #pragma unroll
  for (int off = 16; off > 0; off >>= 1) l += __shfl_xor(l, off, 32);
  if ((threadIdx.x & 31) == 0) atomicAdd(loss, l);
}

// ---------------------------------------------------------------------------
// Kernel 2: separable Gaussian tables in f16.
//   Ex[i*N + n] = exp(-(i - cx_n)^2 / (2 sigma^2))   (H x N, K-major)
//   Ey[j*N + n] = exp(-(j - cy_n)^2 / (2 sigma^2))   (W x N, K-major)
// exp(a+b) = exp(a)*exp(b) turns the 67M-exp splat into Ex * Ey^T (a GEMM).
// ---------------------------------------------------------------------------
__global__ void det_gauss_tables(const float* __restrict__ centers,
                                 const int* __restrict__ stdev_p,
                                 _Float16* __restrict__ Ex,
                                 _Float16* __restrict__ Ey,
                                 int H, int W, int N) {
  int idx = blockIdx.x * blockDim.x + threadIdx.x;
  int total = (H + W) * N;
  if (idx >= total) return;
  float sd = (float)(*stdev_p);
  float inv_denom = 1.0f / (2.0f * sd * sd);
  if (idx < H * N) {
    int i = idx / N, n = idx - i * N;
    float d = (float)i - centers[2 * n + 0];
    Ex[idx] = (_Float16)__expf(-d * d * inv_denom);
  } else {
    int t = idx - H * N;
    int j = t / N, n = t - j * N;
    float d = (float)j - centers[2 * n + 1];
    Ey[t] = (_Float16)__expf(-d * d * inv_denom);
  }
}

// ---------------------------------------------------------------------------
// Fragment loaders per CDNA5 ISA §7.12.2:
//   A (16x32 f16): lane<16 -> row M=lane,    K = k0+0..7  and k0+16..23
//                  lane>=16 -> row M=lane-16, K = k0+8..15 and k0+24..31
//   B (32x16 f16): lane<16 -> col N=lane,    K = k0+0..7  and k0+16..23
//                  lane>=16 -> col N=lane-16, K = k0+8..15 and k0+24..31
// ---------------------------------------------------------------------------
__device__ __forceinline__ v16h load_a_frag(const _Float16* __restrict__ arow,
                                            int k0, int hi) {
  const int abase = k0 + hi * 8;
  v8h lo = *(const v8h*)(arow + abase);        // K = abase .. abase+7
  v8h hh = *(const v8h*)(arow + abase + 16);   // K = abase+16 .. abase+23
  v16h a;
  #pragma unroll
  for (int e = 0; e < 8; ++e) { a[e] = lo[e]; a[e + 8] = hh[e]; }
  return a;
}

__device__ __forceinline__ v16h load_b_frag(const _Float16* __restrict__ brow,
                                            int k0, int hi) {
  const int bbase = k0 + hi * 8;
  v8h lo = *(const v8h*)(brow + bbase);
  v8h hh = *(const v8h*)(brow + bbase + 16);
  v16h b;
  #pragma unroll
  for (int e = 0; e < 8; ++e) { b[e] = lo[e]; b[e + 8] = hh[e]; }
  return b;
}

// ---------------------------------------------------------------------------
// Kernel 3: M_hat = Ex * Ey^T via v_wmma_f32_16x16x32_f16, fused with the
// penalty-reduced focal heatmap loss.
//
// One wave32 computes a 16x32 output tile: one shared A fragment, two B
// fragments, two f32 accumulators. The K-loop is software-pipelined
// (register double-buffering): next K-block's loads are issued before the
// current pair of WMMAs, so s_wait_loadcnt overlaps with matrix work.
// C/D layout: VGPR r -> M = r + 8*(lane>=16), N = lane&15.
// ---------------------------------------------------------------------------
__global__ void __launch_bounds__(32)
det_heatmap_wmma(const _Float16* __restrict__ Ex,
                 const _Float16* __restrict__ Ey,
                 const float* __restrict__ M,
                 float* __restrict__ Mhat,
                 float* __restrict__ loss,
                 int H, int W, int N) {
  const int lane = threadIdx.x & 31;
  const int hi   = lane >> 4;
  const int l15  = lane & 15;
  const int ti   = blockIdx.y * 16;   // output row block (x / H axis)
  const int tj   = blockIdx.x * 32;   // output col block (y / W axis)

  int rowA  = ti + l15;      if (rowA  >= H) rowA  = H - 1;  // clamps keep EXEC full
  int colB0 = tj + l15;      if (colB0 >= W) colB0 = W - 1;
  int colB1 = tj + 16 + l15; if (colB1 >= W) colB1 = W - 1;
  const _Float16* __restrict__ arow  = Ex + (size_t)rowA  * N;
  const _Float16* __restrict__ brow0 = Ey + (size_t)colB0 * N;
  const _Float16* __restrict__ brow1 = Ey + (size_t)colB1 * N;

  v8f acc0 = {}, acc1 = {};

  // Pipeline prologue: fragments for K-block 0.
  v16h a  = load_a_frag(arow, 0, hi);
  v16h b0 = load_b_frag(brow0, 0, hi);
  v16h b1 = load_b_frag(brow1, 0, hi);

  // Steady state: fetch K-block k0+32 while multiplying K-block k0.
  for (int k0 = 0; k0 + 32 < N; k0 += 32) {
    __builtin_prefetch(arow  + k0 + 64, 0, 1);   // global_prefetch_b8, next+1
    __builtin_prefetch(brow0 + k0 + 64, 0, 1);
    __builtin_prefetch(brow1 + k0 + 64, 0, 1);
    v16h an  = load_a_frag(arow, k0 + 32, hi);
    v16h b0n = load_b_frag(brow0, k0 + 32, hi);
    v16h b1n = load_b_frag(brow1, k0 + 32, hi);
    acc0 = __builtin_amdgcn_wmma_f32_16x16x32_f16(false, a, false, b0,
                                                  (short)0, acc0, false, false);
    acc1 = __builtin_amdgcn_wmma_f32_16x16x32_f16(false, a, false, b1,
                                                  (short)0, acc1, false, false);
    a = an; b0 = b0n; b1 = b1n;
  }
  // Epilogue: last K-block.
  acc0 = __builtin_amdgcn_wmma_f32_16x16x32_f16(false, a, false, b0,
                                                (short)0, acc0, false, false);
  acc1 = __builtin_amdgcn_wmma_f32_16x16x32_f16(false, a, false, b1,
                                                (short)0, acc1, false, false);

  // Writeback + fused focal heat loss (alpha = beta = 1).
  float l = 0.0f;
  const int baseRow = ti + hi * 8;
  #pragma unroll
  for (int r = 0; r < 8; ++r) {
    int row = baseRow + r;
    if (row >= H) continue;
    #pragma unroll
    for (int half = 0; half < 2; ++half) {
      int col = tj + half * 16 + l15;
      if (col >= W) continue;
      size_t idx = (size_t)row * W + col;
      float v = half ? acc1[r] : acc0[r];
      Mhat[idx] = v;
      float m  = M[idx];
      float Mh = fminf(fmaxf(v, EPS_F), 1.0f - EPS_F);
      float pos = (1.0f - Mh) * __logf(Mh);
      float neg = (1.0f - m) * Mh * __logf(1.0f - Mh);
      l += (m == 1.0f) ? pos : neg;
    }
  }
  #pragma unroll
  for (int off = 16; off > 0; off >>= 1) l += __shfl_xor(l, off, 32);
  if (lane == 0) atomicAdd(loss, l);
}

// ---------------------------------------------------------------------------
// Host launcher. d_out layout (reference return order, flattened):
//   [0, H*W)            M_hat
//   [H*W]               total loss (L_heat + L_box)
//   [H*W+1, H*W+1+2N)   centers
// d_ws: Ex (H*N f16) followed by Ey (W*N f16).
// ---------------------------------------------------------------------------
extern "C" void kernel_launch(void* const* d_in, const int* in_sizes, int n_in,
                              void* d_out, int out_size, void* d_ws, size_t ws_size,
                              hipStream_t stream) {
  const float* boxes = (const float*)d_in[0];
  const float* M     = (const float*)d_in[1];
  const float* s     = (const float*)d_in[2];
  const float* o     = (const float*)d_in[3];
  const int*   stdev = (const int*)d_in[4];

  const int N  = in_sizes[0] / 4;                 // 256 boxes
  const int HW = in_sizes[1];                     // 1*H*W
  const int H  = (int)(sqrt((double)HW) + 0.5);   // square heatmap (512)
  const int W  = HW / H;

  float* Mhat    = (float*)d_out;
  float* loss    = Mhat + (size_t)HW;
  float* centers = loss + 1;

  _Float16* Ex = (_Float16*)d_ws;
  _Float16* Ey = Ex + (size_t)H * N;

  det_init_loss<<<1, 1, 0, stream>>>(loss);
  det_box_head<<<(N + 255) / 256, 256, 0, stream>>>(boxes, s, o, centers, loss, N);

  const int tableElems = (H + W) * N;
  det_gauss_tables<<<(tableElems + 255) / 256, 256, 0, stream>>>(
      centers, stdev, Ex, Ey, H, W, N);

  dim3 grid((W + 31) / 32, (H + 15) / 16);
  det_heatmap_wmma<<<grid, 32, 0, stream>>>(Ex, Ey, M, Mhat, loss, H, W, N);
}
